// SelfAttention_85796266705174
// MI455X (gfx1250) — matrix operand order repaired and checked by reading, not compiled, not run on hardware
//
#include <hip/hip_runtime.h>

typedef __bf16 bf16;
typedef __attribute__((ext_vector_type(16))) __bf16 v16bf;
typedef __attribute__((ext_vector_type(8)))  float  v8f;

// ---------- WMMA helpers (CDNA5 v_wmma_f32_16x16x32_bf16) ----------

static __device__ __forceinline__ v8f wmma_bf16(v16bf a, v16bf b, v8f c) {
  // (neg_a, A, neg_b, B, c_mod, C, reuse_a, reuse_b)
  return __builtin_amdgcn_wmma_f32_16x16x32_bf16(false, a, false, b, (short)0, c,
                                                 false, false);
}

// A fragment: 16x32 bf16, row-major source, row stride strideM (elements).
// Layout: lanes 0-15 hold M=lane (K 0..15), lanes 16-31 hold M=lane-16
// (K 16..31); VGPR j packs K = koff + 2*j {,+1}.
static __device__ __forceinline__ v16bf load_afrag(const bf16* base, int strideM) {
  const int l    = threadIdx.x & 31;
  const int m    = l & 15;
  const int koff = (l >> 4) * 16;
  const bf16* p  = base + m * strideM;
  v16bf a;
#pragma unroll
  for (int j = 0; j < 8; ++j) {
    const int k0 = koff + 2 * j;
    a[2 * j]     = p[k0];
    a[2 * j + 1] = p[k0 + 1];
  }
  return a;
}

// B fragment: 32x16 bf16 where consecutive K are contiguous in memory for a
// fixed N column; base points at (K=0, N=n0), column stride strideN (elements).
// Layout (same K order as load_afrag): lanes 0-15 = col n, K 0..15; lanes 16-31 = col n, K 16..31.
static __device__ __forceinline__ v16bf load_bfrag(const bf16* base, int strideN) {
  const int l  = threadIdx.x & 31;
  const int n  = l & 15;
  const int kb = (l >> 4) * 16;
  const bf16* p = base + n * strideN + kb;
  v16bf b;
#pragma unroll
  for (int t = 0; t < 16; ++t) b[t] = p[t];
  return b;
}

// ---------- CDNA5 async memory->LDS (ASYNCcnt-tracked, no VGPR staging) ----------

// Per-lane: copy 16 bytes from global gaddr to LDS byte-address lds_addr.
static __device__ __forceinline__ void async_load_b128(unsigned lds_addr,
                                                       const void* gptr) {
  asm volatile("global_load_async_to_lds_b128 %0, %1, off"
               :
               : "v"(lds_addr), "v"((unsigned long long)(uintptr_t)gptr)
               : "memory");
}

#define WAIT_ASYNCCNT_8() asm volatile("s_wait_asynccnt 0x8" ::: "memory")
#define WAIT_ASYNCCNT_0() asm volatile("s_wait_asynccnt 0x0" ::: "memory")

static __device__ __forceinline__ unsigned lds_addr_of(const void* p) {
  // Generic LDS pointer: addr[31:0] is the LDS byte offset (ISA 10.2 aperture map).
  return (unsigned)(uintptr_t)p;
}

// ---------- prep kernels ----------

// x [B,C,HW] f32 -> xT [B,HW,C] bf16   (B=8, C=256, HW=1024)
__global__ void sa_prep_x(const float* __restrict__ x, bf16* __restrict__ xT) {
  const int idx = blockIdx.x * blockDim.x + threadIdx.x;  // (b*C + c)*1024 + p
  const int p = idx & 1023;
  const int c = (idx >> 10) & 255;
  const int b = idx >> 18;
  xT[(b * 1024 + p) * 256 + c] = (bf16)x[idx];
}

// Four 256x256 f32 weight matrices -> bf16 (row-major kept)
__global__ void sa_prep_w(const float* __restrict__ Wq, const float* __restrict__ Wk,
                          const float* __restrict__ Wv, const float* __restrict__ Wm,
                          bf16* __restrict__ out) {
  const int idx = blockIdx.x * blockDim.x + threadIdx.x;  // 0 .. 262143
  const int which = idx >> 16;
  const int off   = idx & 65535;
  float v;
  if      (which == 0) v = Wq[off];
  else if (which == 1) v = Wk[off];
  else if (which == 2) v = Wv[off];
  else                 v = Wm[off];
  out[idx] = (bf16)v;
}

// ---------- QKV projection: Y^T[p,o] = sum_c xT[p,c] * W[o,c] + b[o] ----------
// Q,K -> [B, h, N, hd] bf16 ; V -> [B, h, hd, N] bf16
__global__ __launch_bounds__(128) void sa_qkv_proj(
    const bf16* __restrict__ xT,
    const bf16* __restrict__ Wqb, const bf16* __restrict__ Wkb,
    const bf16* __restrict__ Wvb,
    const float* __restrict__ bq, const float* __restrict__ bk,
    const float* __restrict__ bv,
    bf16* __restrict__ Qh, bf16* __restrict__ Kh, bf16* __restrict__ Vt) {
  const int tile = blockIdx.x * 4 + (threadIdx.x >> 5);  // 8192 tiles
  const int ot = tile & 15;          // output-channel tile (16 ch)
  const int pt = (tile >> 4) & 63;   // pixel tile (16 px)
  const int b  = tile >> 10;

  const bf16* A0 = xT + (b * 1024 + pt * 16) * 256;
  v8f q = {}, k = {}, v = {};
#pragma unroll
  for (int ks = 0; ks < 8; ++ks) {
    const v16bf a  = load_afrag(A0 + ks * 32, 256);
    const v16bf fq = load_bfrag(Wqb + ot * 16 * 256 + ks * 32, 256);
    q = wmma_bf16(a, fq, q);
    const v16bf fk = load_bfrag(Wkb + ot * 16 * 256 + ks * 32, 256);
    k = wmma_bf16(a, fk, k);
    const v16bf fv = load_bfrag(Wvb + ot * 16 * 256 + ks * 32, 256);
    v = wmma_bf16(a, fv, v);
  }
  const int l  = threadIdx.x & 31;
  const int n  = l & 15;
  const int hi = l >> 4;
  const int o  = ot * 16 + n;
  const int head = o >> 5, d = o & 31;
  const float Bq = bq[o], Bk = bk[o], Bv = bv[o];
  const int p0 = pt * 16 + hi * 8;
  bf16* qdst = Qh + (((b * 8 + head) * 1024) + p0) * 32 + d;
  bf16* kdst = Kh + (((b * 8 + head) * 1024) + p0) * 32 + d;
  bf16* vdst = Vt + (((b * 8 + head) * 32) + d) * 1024 + p0;
#pragma unroll
  for (int r = 0; r < 8; ++r) {
    qdst[r * 32] = (bf16)(q[r] + Bq);
    kdst[r * 32] = (bf16)(k[r] + Bk);
    vdst[r]      = (bf16)(v[r] + Bv);
  }
}

// ---------- flash attention: one wave per (b, head, 16-query block) ----------
// K/V tiles for each 32-key step are staged into LDS by the async engine
// (double-buffered, one step in flight), WMMAs consume LDS fragments.

// Issue 8 async b128 ops: K tile (32 keys x 32 d, contiguous 2KB) and
// V tile (32 d-rows x 64B, global row stride 2048B) for keys [j0, j0+32).
static __device__ __forceinline__ void stage_kv_async(const bf16* Kb, const bf16* Vb,
                                                      int j0, bf16* kdst, bf16* vdst) {
  const int l = threadIdx.x & 31;
  {
    const unsigned lds = lds_addr_of(kdst) + l * 16;
    const char* g = (const char*)(Kb + j0 * 32) + l * 16;
#pragma unroll
    for (int i = 0; i < 4; ++i) async_load_b128(lds + i * 512, g + i * 512);
  }
  {
#pragma unroll
    for (int i = 0; i < 4; ++i) {
      const int ci  = i * 32 + l;        // 16B chunk id, 0..127
      const int row = ci >> 2;           // d row 0..31
      const int col = ci & 3;            // chunk within row
      const unsigned lds = lds_addr_of(vdst) + ci * 16;
      const char* g = (const char*)Vb + row * 2048 + j0 * 2 + col * 16;
      async_load_b128(lds, g);
    }
  }
}

__global__ __launch_bounds__(128) void sa_attn(
    const bf16* __restrict__ Qh, const bf16* __restrict__ Kh,
    const bf16* __restrict__ Vt, bf16* __restrict__ AttnT) {
  __shared__ __align__(16) bf16 Plds[4][16][32];       // P bounce, 4KB
  __shared__ __align__(16) bf16 Ktile[4][2][32][32];   // [key][d], dbl-buf, 16KB
  __shared__ __align__(16) bf16 Vtile[4][2][32][32];   // [d][key], dbl-buf, 16KB

  const int wid  = threadIdx.x >> 5;
  const int wave = blockIdx.x * 4 + wid;   // 4096 waves
  const int rb   = wave & 63;              // query row block
  const int head = (wave >> 6) & 7;
  const int b    = wave >> 9;
  const int bh   = b * 8 + head;

  const bf16* Qb = Qh + (bh * 1024 + rb * 16) * 32;
  const bf16* Kb = Kh + bh * 1024 * 32;
  const bf16* Vb = Vt + bh * 32 * 1024;

  const v16bf aq = load_afrag(Qb, 32);   // Q tile constant across the K loop
  v8f O0 = {}, O1 = {};
  float mrow[8], lrow[8];
#pragma unroll
  for (int r = 0; r < 8; ++r) { mrow[r] = -1e30f; lrow[r] = 0.0f; }
  const float scale = 0.17677669529663687f;  // 32^-0.5

  stage_kv_async(Kb, Vb, 0, &Ktile[wid][0][0][0], &Vtile[wid][0][0][0]);

  for (int j = 0; j < 32; ++j) {
    const int buf = j & 1;
    if (j < 31) {
      stage_kv_async(Kb, Vb, (j + 1) * 32,
                     &Ktile[wid][buf ^ 1][0][0], &Vtile[wid][buf ^ 1][0][0]);
      WAIT_ASYNCCNT_8();   // in-order done: current buffer's 8 ops complete
    } else {
      WAIT_ASYNCCNT_0();
    }

    const bf16* kt = &Ktile[wid][buf][0][0];
    const v16bf bk0 = load_bfrag(kt, 32);            // keys 0..15 of the tile
    const v16bf bk1 = load_bfrag(kt + 16 * 32, 32);  // keys 16..31
    v8f z = {};
    v8f S0 = wmma_bf16(aq, bk0, z);
    v8f S1 = wmma_bf16(aq, bk1, z);

#pragma unroll
    for (int r = 0; r < 8; ++r) {
      const float s0 = S0[r] * scale;
      const float s1 = S1[r] * scale;
      float mx = fmaxf(s0, s1);               // row r (+8 on hi half) lives on
      mx = fmaxf(mx, __shfl_xor(mx, 1, 32));  // 16 lanes of one wave half
      mx = fmaxf(mx, __shfl_xor(mx, 2, 32));
      mx = fmaxf(mx, __shfl_xor(mx, 4, 32));
      mx = fmaxf(mx, __shfl_xor(mx, 8, 32));
      const float mnew = fmaxf(mrow[r], mx);
      const float a  = __expf(mrow[r] - mnew);
      const float p0 = __expf(s0 - mnew);
      const float p1 = __expf(s1 - mnew);
      float ps = p0 + p1;
      ps += __shfl_xor(ps, 1, 32);
      ps += __shfl_xor(ps, 2, 32);
      ps += __shfl_xor(ps, 4, 32);
      ps += __shfl_xor(ps, 8, 32);
      lrow[r] = lrow[r] * a + ps;
      mrow[r] = mnew;
      S0[r] = p0; S1[r] = p1;
      O0[r] *= a; O1[r] *= a;
    }

    // D-layout P -> row-major LDS tile
    const int l  = threadIdx.x & 31;
    const int n  = l & 15;
    const int hi = l >> 4;
#pragma unroll
    for (int r = 0; r < 8; ++r) {
      const int m = r + hi * 8;
      Plds[wid][m][n]      = (bf16)S0[r];
      Plds[wid][m][n + 16] = (bf16)S1[r];
    }
    __syncthreads();

    const v16bf ap  = load_afrag(&Plds[wid][0][0], 32);
    const bf16* vt  = &Vtile[wid][buf][0][0];
    const v16bf bv0 = load_bfrag(vt, 32);            // d = 0..15
    const v16bf bv1 = load_bfrag(vt + 16 * 32, 32);  // d = 16..31
    O0 = wmma_bf16(ap, bv0, O0);
    O1 = wmma_bf16(ap, bv1, O1);
    __syncthreads();
  }

  // epilogue: normalize, write channel = d*HEADS + head layout for out-proj
  const int l  = threadIdx.x & 31;
  const int n  = l & 15;
  const int hi = l >> 4;
  const int p0 = rb * 16 + hi * 8;
#pragma unroll
  for (int r = 0; r < 8; ++r) {
    const float inv = 1.0f / lrow[r];
    const int p = p0 + r;
    AttnT[(b * 1024 + p) * 256 + (n * 8 + head)]        = (bf16)(O0[r] * inv);
    AttnT[(b * 1024 + p) * 256 + ((16 + n) * 8 + head)] = (bf16)(O1[r] * inv);
  }
}

// ---------- output projection: y[b,o,p] = sum_c Wm[o,c]*AttnT[b,p,c] + bm[o] ----------
__global__ __launch_bounds__(128) void sa_out_proj(
    const bf16* __restrict__ AttnT, const bf16* __restrict__ Wmb,
    const float* __restrict__ bm, float* __restrict__ out) {
  const int tile = blockIdx.x * 4 + (threadIdx.x >> 5);  // 8192 tiles
  const int ot = tile & 15;
  const int pt = (tile >> 4) & 63;
  const int b  = tile >> 10;
  const bf16* A0 = AttnT + (b * 1024 + pt * 16) * 256;
  v8f acc = {};
#pragma unroll
  for (int ks = 0; ks < 8; ++ks) {
    const v16bf a  = load_afrag(A0 + ks * 32, 256);
    const v16bf bw = load_bfrag(Wmb + ot * 16 * 256 + ks * 32, 256);
    acc = wmma_bf16(a, bw, acc);
  }
  const int l  = threadIdx.x & 31;
  const int n  = l & 15;
  const int hi = l >> 4;
  const int o  = ot * 16 + n;
  const float bias = bm[o];
  const int p0 = pt * 16 + hi * 8;
  float* dst = out + (b * 256 + o) * 1024 + p0;
#pragma unroll
  for (int r = 0; r < 8; ++r) dst[r] = acc[r] + bias;  // 32B contiguous per lane
}

// ---------- launch ----------
extern "C" void kernel_launch(void* const* d_in, const int* in_sizes, int n_in,
                              void* d_out, int out_size, void* d_ws, size_t ws_size,
                              hipStream_t stream) {
  const float* x  = (const float*)d_in[0];
  const float* Wq = (const float*)d_in[1];
  const float* bq = (const float*)d_in[2];
  const float* Wk = (const float*)d_in[3];
  const float* bk = (const float*)d_in[4];
  const float* Wv = (const float*)d_in[5];
  const float* bv = (const float*)d_in[6];
  const float* Wm = (const float*)d_in[7];
  const float* bm = (const float*)d_in[8];
  float* out = (float*)d_out;

  char* ws = (char*)d_ws;
  bf16* xT   = (bf16*)(ws + 0);           // 4 MB
  bf16* Wall = (bf16*)(ws + 4194304);     // 512 KB (4 x 256x256 bf16)
  bf16* Qh   = (bf16*)(ws + 4718592);     // 4 MB  [B,h,N,hd]
  bf16* Kh   = (bf16*)(ws + 8912896);     // 4 MB  [B,h,N,hd]
  bf16* Vt   = (bf16*)(ws + 13107200);    // 4 MB  [B,h,hd,N]
  bf16* At   = (bf16*)(ws + 17301504);    // 4 MB  [B,N,C]
  bf16* Wqb = Wall;
  bf16* Wkb = Wall + 65536;
  bf16* Wvb = Wall + 131072;
  bf16* Wmb = Wall + 196608;

  sa_prep_x<<<8192, 256, 0, stream>>>(x, xT);
  sa_prep_w<<<1024, 256, 0, stream>>>(Wq, Wk, Wv, Wm, Wall);
  sa_qkv_proj<<<2048, 128, 0, stream>>>(xT, Wqb, Wkb, Wvb, bq, bk, bv, Qh, Kh, Vt);
  sa_attn<<<1024, 128, 0, stream>>>(Qh, Kh, Vt, At);
  sa_out_proj<<<2048, 128, 0, stream>>>(At, Wmb, bm, out);
}
